// SelfAttention_13795434955363
// MI455X (gfx1250) — compile-verified, matrix operand order checked by reading, not run on hardware
//
#include <hip/hip_runtime.h>

// ---------------------------------------------------------------------------
// CDNA5 (gfx1250) bf16 WMMA self-attention.
// D = 1024, H = 16, Dh = 64, L = 4096, B = 1.
// ---------------------------------------------------------------------------

typedef __attribute__((ext_vector_type(16))) __bf16 bf16x16;
typedef __attribute__((ext_vector_type(8)))  __bf16 bf16x8;
typedef __attribute__((ext_vector_type(8)))  float  f32x8;

union ABFrag {
    bf16x16 v;
    bf16x8  h[2];
};

// Load an A/B fragment: two contiguous 8-element (16B) bf16 chunks.
__device__ inline bf16x16 load_ab(const __bf16* p0, const __bf16* p1) {
    ABFrag f;
    f.h[0] = *(const bf16x8*)p0;
    f.h[1] = *(const bf16x8*)p1;
    return f.v;
}

__device__ inline f32x8 wmma_bf16(bf16x16 a, bf16x16 b, f32x8 c) {
    // (neg_a, A, neg_b, B, c_mod, C, reuse_a, reuse_b)
    return __builtin_amdgcn_wmma_f32_16x16x32_bf16(false, a, false, b,
                                                   (short)0, c, false, false);
}

// 32x64 output tile (2x4 WMMA fragments), C = A[M,K] @ B[N,K]^T, bf16 inputs.
// 8 wmma per K-step of 32 against 12 b128 loads.
__device__ inline void gemm32x64(const __bf16* __restrict__ A,
                                 const __bf16* __restrict__ Bm,
                                 int K, int rowA, int rowB,
                                 f32x8 acc[2][4]) {
    const int lane = threadIdx.x & 31;
    const int l16  = lane & 15;
    const int half = lane >> 4;
    const int off0 = half * 8;
    const int off1 = 16 + half * 8;
    const __bf16* a0 = A + (size_t)(rowA      + l16) * K;
    const __bf16* a1 = A + (size_t)(rowA + 16 + l16) * K;
    const __bf16* b0 = Bm + (size_t)(rowB      + l16) * K;
    const __bf16* b1 = Bm + (size_t)(rowB + 16 + l16) * K;
    const __bf16* b2 = Bm + (size_t)(rowB + 32 + l16) * K;
    const __bf16* b3 = Bm + (size_t)(rowB + 48 + l16) * K;
    #pragma unroll 2
    for (int k0 = 0; k0 < K; k0 += 32) {
        bf16x16 fa0 = load_ab(a0 + k0 + off0, a0 + k0 + off1);
        bf16x16 fa1 = load_ab(a1 + k0 + off0, a1 + k0 + off1);
        bf16x16 fb;
        fb = load_ab(b0 + k0 + off0, b0 + k0 + off1);
        acc[0][0] = wmma_bf16(fa0, fb, acc[0][0]);
        acc[1][0] = wmma_bf16(fa1, fb, acc[1][0]);
        fb = load_ab(b1 + k0 + off0, b1 + k0 + off1);
        acc[0][1] = wmma_bf16(fa0, fb, acc[0][1]);
        acc[1][1] = wmma_bf16(fa1, fb, acc[1][1]);
        fb = load_ab(b2 + k0 + off0, b2 + k0 + off1);
        acc[0][2] = wmma_bf16(fa0, fb, acc[0][2]);
        acc[1][2] = wmma_bf16(fa1, fb, acc[1][2]);
        fb = load_ab(b3 + k0 + off0, b3 + k0 + off1);
        acc[0][3] = wmma_bf16(fa0, fb, acc[0][3]);
        acc[1][3] = wmma_bf16(fa1, fb, acc[1][3]);
    }
}

// ---------------------------------------------------------------------------
// Kernel 1: fp32 -> bf16 conversion (4 elems / thread, n % 1024 == 0).
// ---------------------------------------------------------------------------
__global__ __launch_bounds__(256) void cvt_bf16_kernel(
        const float* __restrict__ src, __bf16* __restrict__ dst, int n) {
    int i = (blockIdx.x * 256 + threadIdx.x) * 4;
    if (i + 3 < n) {
        float4 f = *(const float4*)(src + i);
        dst[i + 0] = (__bf16)f.x;
        dst[i + 1] = (__bf16)f.y;
        dst[i + 2] = (__bf16)f.z;
        dst[i + 3] = (__bf16)f.w;
    }
}

// ---------------------------------------------------------------------------
// Kernel 2: QKV = x @ W_qkv^T, scattered into attention layouts:
//   Q, K : [h][l][64]   (row-major rows for A/B fragments of Q@K^T)
//   V    : [h][64][L]   (transposed so PV B-fragments load contiguously)
// ---------------------------------------------------------------------------
__global__ __launch_bounds__(256) void qkv_kernel(
        const __bf16* __restrict__ xb, const __bf16* __restrict__ wb,
        __bf16* __restrict__ Qws, __bf16* __restrict__ Kws,
        __bf16* __restrict__ Vws) {
    const int K  = 1024;
    const int NT = 3072 / 64;
    int wid = blockIdx.x * 8 + (threadIdx.x >> 5);
    int tm = wid / NT;
    int tn = wid % NT;
    f32x8 acc[2][4] = {};
    gemm32x64(xb, wb, K, tm * 32, tn * 64, acc);

    const int lane = threadIdx.x & 31;
    const int l16  = lane & 15;
    const int half = lane >> 4;
    #pragma unroll
    for (int mi = 0; mi < 2; ++mi) {
        #pragma unroll
        for (int ni = 0; ni < 4; ++ni) {
            int n   = tn * 64 + ni * 16 + l16;    // column in [0, 3072)
            int sec = n >> 10;                    // 0=Q, 1=K, 2=V
            int cin = n & 1023;
            int hh  = cin >> 6;
            int dd  = cin & 63;
            int lb  = tm * 32 + mi * 16 + half * 8;
            f32x8 c = acc[mi][ni];
            #pragma unroll
            for (int r = 0; r < 8; ++r) {
                int l = lb + r;
                __bf16 v = (__bf16)c[r];
                if (sec == 0)      Qws[((size_t)hh * 4096 + l) * 64 + dd] = v;
                else if (sec == 1) Kws[((size_t)hh * 4096 + l) * 64 + dd] = v;
                else               Vws[((size_t)hh * 64 + dd) * 4096 + l] = v;
            }
        }
    }
}

// ---------------------------------------------------------------------------
// Kernel 3: flash attention. One wave = (head, 16-query tile), Bc = 64 keys.
// Per key block: 8 wmma for S = Q@K^T, 8 wmma for P@V.
// ---------------------------------------------------------------------------
__global__ __launch_bounds__(256) void attn_kernel(
        const __bf16* __restrict__ Qws, const __bf16* __restrict__ Kws,
        const __bf16* __restrict__ Vws, __bf16* __restrict__ ctx) {
    __shared__ __bf16 Pbuf[8][16 * 64];   // per-wave 16x64 P tile (16 KB)

    const int L  = 4096;
    const int Dh = 64;
    int wid = blockIdx.x * 8 + (threadIdx.x >> 5);   // 0..4095
    int hh  = wid >> 8;                              // head
    int qt  = wid & 255;                             // 16-query tile
    const int lane = threadIdx.x & 31;
    const int l16  = lane & 15;
    const int half = lane >> 4;
    const int off0 = half * 8;
    const int off1 = 16 + half * 8;

    // Q fragments (Dh = 64 -> two k-steps of 32)
    const __bf16* Qb = Qws + ((size_t)hh * L + qt * 16 + l16) * Dh;
    bf16x16 qa0 = load_ab(Qb + off0,      Qb + off1);
    bf16x16 qa1 = load_ab(Qb + 32 + off0, Qb + 32 + off1);

    const __bf16* Kb = Kws + (size_t)hh * L * Dh;
    const __bf16* Vb = Vws + (size_t)hh * Dh * L;

    f32x8 ctxacc[4] = {};
    float mrow[8], lrow[8];
    #pragma unroll
    for (int r = 0; r < 8; ++r) { mrow[r] = -1e30f; lrow[r] = 0.0f; }

    __bf16* Pw = Pbuf[threadIdx.x >> 5];

    for (int kb = 0; kb < L; kb += 64) {
        // ---- S = (Q @ K^T) * 1/sqrt(Dh), 16x64 in four C fragments ----
        f32x8 s[4];
        #pragma unroll
        for (int j = 0; j < 4; ++j) {
            const __bf16* kp = Kb + (size_t)(kb + j * 16 + l16) * Dh;
            f32x8 sj = {};
            sj = wmma_bf16(qa0, load_ab(kp + off0,      kp + off1),      sj);
            sj = wmma_bf16(qa1, load_ab(kp + 32 + off0, kp + 32 + off1), sj);
            s[j] = sj;
        }

        // ---- online softmax; row m = r + half*8 spans 16 lanes of a half ----
        #pragma unroll
        for (int r = 0; r < 8; ++r) {
            float v0 = s[0][r] * 0.125f;
            float v1 = s[1][r] * 0.125f;
            float v2 = s[2][r] * 0.125f;
            float v3 = s[3][r] * 0.125f;
            float mx = fmaxf(fmaxf(v0, v1), fmaxf(v2, v3));
            mx = fmaxf(mx, __shfl_xor(mx, 1));
            mx = fmaxf(mx, __shfl_xor(mx, 2));
            mx = fmaxf(mx, __shfl_xor(mx, 4));
            mx = fmaxf(mx, __shfl_xor(mx, 8));
            float mnew  = fmaxf(mrow[r], mx);
            float alpha = __expf(mrow[r] - mnew);
            float p0 = __expf(v0 - mnew);
            float p1 = __expf(v1 - mnew);
            float p2 = __expf(v2 - mnew);
            float p3 = __expf(v3 - mnew);
            float rs = (p0 + p1) + (p2 + p3);
            rs += __shfl_xor(rs, 1);
            rs += __shfl_xor(rs, 2);
            rs += __shfl_xor(rs, 4);
            rs += __shfl_xor(rs, 8);
            lrow[r] = lrow[r] * alpha + rs;
            mrow[r] = mnew;
            int m = r + half * 8;
            Pw[m * 64 +      l16] = (__bf16)p0;
            Pw[m * 64 + 16 + l16] = (__bf16)p1;
            Pw[m * 64 + 32 + l16] = (__bf16)p2;
            Pw[m * 64 + 48 + l16] = (__bf16)p3;
            #pragma unroll
            for (int f = 0; f < 4; ++f) ctxacc[f][r] *= alpha;
        }

        // ---- C-layout -> A-layout via wave-private LDS, then P @ V ----
        bf16x16 pA0 = load_ab(&Pw[l16 * 64 + off0],      &Pw[l16 * 64 + off1]);
        bf16x16 pA1 = load_ab(&Pw[l16 * 64 + 32 + off0], &Pw[l16 * 64 + 32 + off1]);
        #pragma unroll
        for (int nc = 0; nc < 4; ++nc) {
            const __bf16* vp = Vb + (size_t)(nc * 16 + l16) * L + kb;
            ctxacc[nc] = wmma_bf16(pA0, load_ab(vp + off0, vp + off1),
                                   ctxacc[nc]);
            ctxacc[nc] = wmma_bf16(pA1, load_ab(vp + 32 + off0, vp + 32 + off1),
                                   ctxacc[nc]);
        }
    }

    // ---- normalize and store ctx as [l][1024] bf16 ----
    int qbase = qt * 16 + half * 8;
    #pragma unroll
    for (int nc = 0; nc < 4; ++nc) {
        #pragma unroll
        for (int r = 0; r < 8; ++r) {
            float v = ctxacc[nc][r] / lrow[r];
            ctx[(size_t)(qbase + r) * 1024 + hh * 64 + nc * 16 + l16] = (__bf16)v;
        }
    }
}

// ---------------------------------------------------------------------------
// Kernel 4: out = ctx @ W_out^T, fp32 output.
// ---------------------------------------------------------------------------
__global__ __launch_bounds__(256) void out_kernel(
        const __bf16* __restrict__ ctxb, const __bf16* __restrict__ wo,
        float* __restrict__ out) {
    const int K  = 1024;
    const int NT = 1024 / 64;
    int wid = blockIdx.x * 8 + (threadIdx.x >> 5);
    int tm = wid / NT;
    int tn = wid % NT;
    f32x8 acc[2][4] = {};
    gemm32x64(ctxb, wo, K, tm * 32, tn * 64, acc);

    const int lane = threadIdx.x & 31;
    const int l16  = lane & 15;
    const int half = lane >> 4;
    #pragma unroll
    for (int mi = 0; mi < 2; ++mi) {
        #pragma unroll
        for (int ni = 0; ni < 4; ++ni) {
            int n  = tn * 64 + ni * 16 + l16;
            int lb = tm * 32 + mi * 16 + half * 8;
            #pragma unroll
            for (int r = 0; r < 8; ++r)
                out[(size_t)(lb + r) * 1024 + n] = acc[mi][ni][r];
        }
    }
}

// ---------------------------------------------------------------------------
extern "C" void kernel_launch(void* const* d_in, const int* in_sizes, int n_in,
                              void* d_out, int out_size, void* d_ws, size_t ws_size,
                              hipStream_t stream) {
    (void)in_sizes; (void)n_in; (void)out_size; (void)ws_size;
    const float* x    = (const float*)d_in[0];
    const float* Wqkv = (const float*)d_in[1];
    const float* Wout = (const float*)d_in[2];

    char* ws = (char*)d_ws;
    const size_t MB = 1u << 20;
    __bf16* xb   = (__bf16*)(ws + 0 * MB);    // 8 MiB  : x bf16        [4096][1024]
    __bf16* wqb  = (__bf16*)(ws + 8 * MB);    // 6 MiB  : W_qkv bf16    [3072][1024]
    __bf16* wob  = (__bf16*)(ws + 14 * MB);   // 2 MiB  : W_out bf16    [1024][1024]
    __bf16* Qws  = (__bf16*)(ws + 16 * MB);   // 8 MiB  : Q  [16][4096][64]
    __bf16* Kws  = (__bf16*)(ws + 24 * MB);   // 8 MiB  : K  [16][4096][64]
    __bf16* Vws  = (__bf16*)(ws + 32 * MB);   // 8 MiB  : V^T [16][64][4096]
    __bf16* ctxb = (__bf16*)(ws + 40 * MB);   // 8 MiB  : ctx [4096][1024]

    cvt_bf16_kernel<<<4096, 256, 0, stream>>>(x,    xb,  4096 * 1024);
    cvt_bf16_kernel<<<3072, 256, 0, stream>>>(Wqkv, wqb, 3072 * 1024);
    cvt_bf16_kernel<<<1024, 256, 0, stream>>>(Wout, wob, 1024 * 1024);

    // (4096/32) * (3072/64) = 6144 waves / 8 per block
    qkv_kernel<<<768, 256, 0, stream>>>(xb, wqb, Qws, Kws, Vws);
    // 16 heads * 256 query tiles = 4096 waves / 8 per block
    attn_kernel<<<512, 256, 0, stream>>>(Qws, Kws, Vws, ctxb);
    // (4096/32) * (1024/64) = 2048 waves / 8 per block
    out_kernel<<<256, 256, 0, stream>>>(ctxb, wob, (float*)d_out);
}
